// TemporalCrossAttention_52175262711850
// MI455X (gfx1250) — compile-verified, matrix operand order checked by reading, not run on hardware
//
#include <hip/hip_runtime.h>
#include <hip/hip_fp16.h>

// ---------------------------------------------------------------------------
// TemporalCrossAttention for MI455X (gfx1250, wave32, WMMA, async-to-LDS)
// Pipeline: LN(x)->f16, LN(xf)->f16, W->f16 transposed, WMMA GEMMs for Q/K/V
// (double-buffered async global->LDS staging), fused attention (QK^T +
// softmax + PV) per (b,h,128-row tile), emb MLP, AdaLN stylize, final WMMA
// GEMM with residual epilogue.
// ---------------------------------------------------------------------------

typedef _Float16 half_t;
typedef __attribute__((ext_vector_type(16))) _Float16 v16h;
typedef __attribute__((ext_vector_type(8)))  _Float16 h8;
typedef __attribute__((ext_vector_type(8)))  float    v8f;

#define B_  4
#define T_  4096
#define D_  1024
#define N_  256
#define L_  768
#define H_  16
#define DH_ 64
#define TE_ 1024

__device__ inline v8f zero8() {
  v8f z;
#pragma unroll
  for (int i = 0; i < 8; i++) z[i] = 0.f;
  return z;
}

__device__ inline v8f wmma_f16(v16h a, v16h b, v8f c) {
  return __builtin_amdgcn_wmma_f32_16x16x32_f16(
      /*neg_a=*/false, a, /*neg_b=*/false, b,
      /*c_mod=*/(short)0, c, /*reuse_a=*/false, /*reuse_b=*/false);
}

// --- CDNA5 async global->LDS (ASYNCcnt) ------------------------------------
// LDS offset for the VDST operand: flat LDS aperture keeps the wave-relative
// byte offset in addr[31:0].
__device__ inline uint32_t lds_off(const void* p) {
  return (uint32_t)(uintptr_t)p;
}
__device__ inline void async_ld_b128(uint32_t ldsaddr, const void* g) {
  asm volatile("global_load_async_to_lds_b128 %0, %1, off"
               :: "v"(ldsaddr), "v"((unsigned long long)(uintptr_t)g)
               : "memory");
}
__device__ inline void wait_async0() {
  asm volatile("s_wait_asynccnt 0x0" ::: "memory");
}

// Load a 16x32 (MxK) f16 fragment in the ISA A/B layout from a row-major tile:
// lane = (m & 15); K chunks at (lane>>4)*8 and 16+(lane>>4)*8, each 8 halfs.
// tileBase must be 16B aligned and stride a multiple of 8 halfs.
__device__ inline v16h ld_frag(const half_t* tileBase, int stride, int lane) {
  int m  = lane & 15;
  int hi = (lane >> 4) * 8;
  const half_t* p = tileBase + m * stride + hi;
  union { v16h v; h8 h[2]; } u;
  u.h[0] = *(const h8*)(p);
  u.h[1] = *(const h8*)(p + 16);
  return u.v;
}

// ---------------------------------------------------------------------------
// LayerNorm over `width` fp32 -> fp16 row output. One block per row.
// ---------------------------------------------------------------------------
__global__ __launch_bounds__(256)
void ln_rows_f16(const float* __restrict__ in, const float* __restrict__ g,
                 const float* __restrict__ bt, half_t* __restrict__ out, int width) {
  int row = blockIdx.x;
  const float* x = in + (size_t)row * width;
  int tid = threadIdx.x;
  float s = 0.f, s2 = 0.f;
  for (int i = tid; i < width; i += 256) { float v = x[i]; s += v; s2 += v * v; }
#pragma unroll
  for (int m = 16; m; m >>= 1) { s += __shfl_xor(s, m, 32); s2 += __shfl_xor(s2, m, 32); }
  __shared__ float ws[8], ws2[8];
  __shared__ float mu_s, rs_s;
  int w = tid >> 5;
  if ((tid & 31) == 0) { ws[w] = s; ws2[w] = s2; }
  __syncthreads();
  if (tid == 0) {
    float S = 0.f, S2 = 0.f;
#pragma unroll
    for (int i = 0; i < 8; i++) { S += ws[i]; S2 += ws2[i]; }
    float mu = S / (float)width;
    float var = S2 / (float)width - mu * mu;
    mu_s = mu; rs_s = rsqrtf(var + 1e-5f);
  }
  __syncthreads();
  float mu = mu_s, rs = rs_s;
  for (int i = tid; i < width; i += 256)
    out[(size_t)row * width + i] = (half_t)((x[i] - mu) * rs * g[i] + bt[i]);
}

// ---------------------------------------------------------------------------
// Convert fp32 W (K x N, row-major) -> fp16 Wt (N x K) so B-fragments read
// contiguously (B fed column-major to WMMA).
// ---------------------------------------------------------------------------
__global__ __launch_bounds__(256)
void w_to_f16_t(const float* __restrict__ W, half_t* __restrict__ Wt, int K, int N) {
  int total = K * N;
  for (int idx = blockIdx.x * 256 + threadIdx.x; idx < total; idx += gridDim.x * 256) {
    int k = idx / N, n = idx % N;  // coalesced read over n
    Wt[(size_t)n * K + k] = (half_t)W[idx];
  }
}

// ---------------------------------------------------------------------------
// WMMA GEMM: out = alpha*(A[MxK] @ Wt^T + bias) (+ resid for RESID).
// A row-major f16, Wt row-major f16 (N x K). Block = 256 thr = 8 waves,
// tile 128x128x32; wave computes 64x32 (4x2 WMMA tiles).
// Double-buffered LDS, staged with global_load_async_to_lds_b128.
// ---------------------------------------------------------------------------
template <bool RESID>
__global__ __launch_bounds__(256)
void gemm_wmma(const half_t* __restrict__ A, const half_t* __restrict__ Wt,
               const float* __restrict__ bias, const float* __restrict__ resid,
               half_t* __restrict__ outh, float* __restrict__ outf,
               int M, int Nd, int Kd, float alpha) {
  __shared__ __align__(16) half_t As[2][128 * 40];
  __shared__ __align__(16) half_t Bs[2][128 * 40];
  int tid = threadIdx.x, lane = tid & 31, w = tid >> 5;
  int wm = w & 1, wn = w >> 1;
  int m0 = blockIdx.y * 128, n0 = blockIdx.x * 128;

  v8f acc[4][2];
#pragma unroll
  for (int i = 0; i < 4; i++)
#pragma unroll
    for (int j = 0; j < 2; j++) acc[i][j] = zero8();

  int r  = tid >> 1;        // staging row (128 rows, 2 threads/row)
  int cH = (tid & 1) * 16;  // staging half-chunk (16 halfs)
  const half_t* gA = A  + (size_t)(m0 + r) * Kd + cH;
  const half_t* gB = Wt + (size_t)(n0 + r) * Kd + cH;
  uint32_t lA[2], lB[2];
#pragma unroll
  for (int s = 0; s < 2; s++) {
    lA[s] = lds_off(&As[s][r * 40 + cH]);
    lB[s] = lds_off(&Bs[s][r * 40 + cH]);
  }

  auto issue = [&](int s, int k0) {
    async_ld_b128(lA[s],      gA + k0);
    async_ld_b128(lA[s] + 16, gA + k0 + 8);
    async_ld_b128(lB[s],      gB + k0);
    async_ld_b128(lB[s] + 16, gB + k0 + 8);
  };

  issue(0, 0);
  int cur = 0;
  for (int k0 = 0; k0 < Kd; k0 += 32, cur ^= 1) {
    wait_async0();    // own async writes done
    __syncthreads();  // everyone's writes visible; prev readers of buf cur^1 done
    if (k0 + 32 < Kd) issue(cur ^ 1, k0 + 32);

    v16h af[4], bf[2];
#pragma unroll
    for (int i = 0; i < 4; i++)
      af[i] = ld_frag(&As[cur][(wm * 64 + i * 16) * 40], 40, lane);
#pragma unroll
    for (int j = 0; j < 2; j++)
      bf[j] = ld_frag(&Bs[cur][(wn * 32 + j * 16) * 40], 40, lane);
#pragma unroll
    for (int i = 0; i < 4; i++)
#pragma unroll
      for (int j = 0; j < 2; j++) acc[i][j] = wmma_f16(af[i], bf[j], acc[i][j]);
  }

  // Epilogue: C layout -> VGPR v: row v (lanes 0-15) / v+8 (lanes 16-31)
  int mrow = m0 + wm * 64;
  int nc0  = n0 + wn * 32;
#pragma unroll
  for (int i = 0; i < 4; i++)
#pragma unroll
    for (int j = 0; j < 2; j++) {
      int n = nc0 + j * 16 + (lane & 15);
      float bn = bias[n];
#pragma unroll
      for (int v = 0; v < 8; v++) {
        int row = mrow + i * 16 + v + ((lane >> 4) << 3);
        float val = (acc[i][j][v] + bn) * alpha;
        if (RESID)
          outf[(size_t)row * Nd + n] = val + resid[(size_t)row * Nd + n];
        else
          outh[(size_t)row * Nd + n] = (half_t)val;
      }
    }
}

// ---------------------------------------------------------------------------
// Fused attention per (b,h) and 128-row Q tile. K (256x64) staged to LDS
// via async-to-LDS; V staged transposed (64x256). S = QK^T via WMMA (Q
// pre-scaled by 1/8 in the Q GEMM), exact softmax over N=256 (16 lanes x 16
// tiles, shfl_xor reduction within half-wave), P->LDS f16, y = P@V via WMMA.
// ---------------------------------------------------------------------------
__global__ __launch_bounds__(256)
void attn_kernel(const half_t* __restrict__ qh, const half_t* __restrict__ kh,
                 const half_t* __restrict__ vh, half_t* __restrict__ y) {
  __shared__ __align__(16) half_t Ks[256 * 72];   // 36 KB
  __shared__ __align__(16) half_t Vt[64 * 264];   // 33 KB (V transposed)
  __shared__ __align__(16) half_t Ps[128 * 264];  // 66 KB
  int bh = blockIdx.x;
  int b = bh / H_, h = bh % H_;
  int t0 = blockIdx.y * 128;
  int tid = threadIdx.x, lane = tid & 31, w = tid >> 5;

  // Stage K (async global->LDS) and V^T (manual transpose)
  for (int c = tid; c < 2048; c += 256) {
    int tok = c >> 3, d0 = (c & 7) * 8;
    size_t gbase = ((size_t)(b * N_ + tok)) * D_ + h * DH_ + d0;
    async_ld_b128(lds_off(Ks + tok * 72 + d0), kh + gbase);
    h8 vv = *(const h8*)(vh + gbase);
#pragma unroll
    for (int e = 0; e < 8; e++) Vt[(d0 + e) * 264 + tok] = vv[e];
  }

  // Q fragments for this wave's 16 rows (dh=64 -> two K=32 fragments)
  v16h qa[2];
  {
    int m = lane & 15, hi = (lane >> 4) * 8;
    int t = t0 + w * 16 + m;
    const half_t* qp = qh + ((size_t)(b * T_ + t)) * D_ + h * DH_;
#pragma unroll
    for (int ks = 0; ks < 2; ks++) {
      union { v16h v; h8 hh[2]; } u;
      u.hh[0] = *(const h8*)(qp + ks * 32 + hi);
      u.hh[1] = *(const h8*)(qp + ks * 32 + 16 + hi);
      qa[ks] = u.v;
    }
  }
  wait_async0();
  __syncthreads();

  // S = Q K^T : 16 column tiles of 16, accumulate over dh in 2 steps
  v8f accs[16];
#pragma unroll
  for (int nt = 0; nt < 16; nt++) {
    accs[nt] = zero8();
#pragma unroll
    for (int ks = 0; ks < 2; ks++) {
      v16h bf = ld_frag(Ks + (nt * 16) * 72 + ks * 32, 72, lane);
      accs[nt] = wmma_f16(qa[ks], bf, accs[nt]);
    }
  }

  // Softmax over 256 cols: per C-VGPR j the row is w*16 + j + 8*(lane>=16);
  // 16 values live locally (one per tile), 16 more across lanes of half-wave.
  float rinv[8];
#pragma unroll
  for (int j = 0; j < 8; j++) {
    float m = -1e30f;
#pragma unroll
    for (int nt = 0; nt < 16; nt++) m = fmaxf(m, accs[nt][j]);
#pragma unroll
    for (int msk = 8; msk; msk >>= 1) m = fmaxf(m, __shfl_xor(m, msk, 32));
    float ssum = 0.f;
#pragma unroll
    for (int nt = 0; nt < 16; nt++) {
      float e = __expf(accs[nt][j] - m);
      accs[nt][j] = e;
      ssum += e;
    }
#pragma unroll
    for (int msk = 8; msk; msk >>= 1) ssum += __shfl_xor(ssum, msk, 32);
    rinv[j] = 1.f / ssum;
  }

  // P -> LDS (each wave owns its 16 rows; same-wave LDS RAW is in-order)
#pragma unroll
  for (int nt = 0; nt < 16; nt++)
#pragma unroll
    for (int j = 0; j < 8; j++) {
      int rl = w * 16 + j + ((lane >> 4) << 3);
      Ps[rl * 264 + nt * 16 + (lane & 15)] = (half_t)(accs[nt][j] * rinv[j]);
    }

  // y = P @ V : K=256 in 8 steps of 32; 4 output tiles (dh=64)
  v8f accy[4];
#pragma unroll
  for (int j = 0; j < 4; j++) accy[j] = zero8();
  for (int kt = 0; kt < 8; kt++) {
    v16h af = ld_frag(Ps + (w * 16) * 264 + kt * 32, 264, lane);
#pragma unroll
    for (int j = 0; j < 4; j++) {
      v16h bf = ld_frag(Vt + (j * 16) * 264 + kt * 32, 264, lane);
      accy[j] = wmma_f16(af, bf, accy[j]);
    }
  }

#pragma unroll
  for (int j = 0; j < 4; j++) {
    int n = h * DH_ + j * 16 + (lane & 15);
#pragma unroll
    for (int v = 0; v < 8; v++) {
      int t = t0 + w * 16 + v + ((lane >> 4) << 3);
      y[((size_t)(b * T_ + t)) * D_ + n] = (half_t)accy[j][v];
    }
  }
}

// ---------------------------------------------------------------------------
// emb_out[b, o] = silu(emb[b]) @ We + be   (B=4 x 2048, tiny)
// ---------------------------------------------------------------------------
__global__ __launch_bounds__(256)
void emb_out_kernel(const float* __restrict__ emb, const float* __restrict__ We,
                    const float* __restrict__ be, float* __restrict__ eo) {
  int b = blockIdx.y;
  int o = blockIdx.x * 256 + threadIdx.x;  // 0..2047
  __shared__ float se[TE_];
  for (int i = threadIdx.x; i < TE_; i += 256) {
    float v = emb[b * TE_ + i];
    se[i] = v / (1.f + __expf(-v));
  }
  __syncthreads();
  float acc = 0.f;
  for (int i = 0; i < TE_; i++) acc += se[i] * We[(size_t)i * (2 * D_) + o];
  eo[b * (2 * D_) + o] = acc + be[o];
}

// ---------------------------------------------------------------------------
// Stylize: h = silu( LN(y) * (1 + scale_b) + shift_b ) -> f16
// ---------------------------------------------------------------------------
__global__ __launch_bounds__(256)
void stylize_kernel(const half_t* __restrict__ y, const float* __restrict__ eo,
                    const float* __restrict__ g, const float* __restrict__ bt,
                    half_t* __restrict__ out) {
  int row = blockIdx.x;  // b*T + t
  int b = row / T_;
  const half_t* yr = y + (size_t)row * D_;
  int tid = threadIdx.x;
  float s = 0.f, s2 = 0.f;
  for (int i = tid; i < D_; i += 256) { float v = (float)yr[i]; s += v; s2 += v * v; }
#pragma unroll
  for (int m = 16; m; m >>= 1) { s += __shfl_xor(s, m, 32); s2 += __shfl_xor(s2, m, 32); }
  __shared__ float ws[8], ws2[8];
  __shared__ float mu_s, rs_s;
  int w = tid >> 5;
  if ((tid & 31) == 0) { ws[w] = s; ws2[w] = s2; }
  __syncthreads();
  if (tid == 0) {
    float S = 0.f, S2 = 0.f;
#pragma unroll
    for (int i = 0; i < 8; i++) { S += ws[i]; S2 += ws2[i]; }
    float mu = S / (float)D_;
    float var = S2 / (float)D_ - mu * mu;
    mu_s = mu; rs_s = rsqrtf(var + 1e-5f);
  }
  __syncthreads();
  float mu = mu_s, rs = rs_s;
  for (int i = tid; i < D_; i += 256) {
    float ln = ((float)yr[i] - mu) * rs * g[i] + bt[i];
    float sc = eo[b * (2 * D_) + i];
    float sh = eo[b * (2 * D_) + D_ + i];
    float hv = ln * (1.f + sc) + sh;
    hv = hv / (1.f + __expf(-hv));  // silu
    out[(size_t)row * D_ + i] = (half_t)hv;
  }
}

// ---------------------------------------------------------------------------
extern "C" void kernel_launch(void* const* d_in, const int* in_sizes, int n_in,
                              void* d_out, int out_size, void* d_ws, size_t ws_size,
                              hipStream_t stream) {
  (void)in_sizes; (void)n_in; (void)out_size; (void)ws_size;
  const float* x       = (const float*)d_in[0];
  const float* xf      = (const float*)d_in[1];
  const float* emb     = (const float*)d_in[2];
  const float* norm_g  = (const float*)d_in[3];
  const float* norm_b  = (const float*)d_in[4];
  const float* tnorm_g = (const float*)d_in[5];
  const float* tnorm_b = (const float*)d_in[6];
  const float* Wq      = (const float*)d_in[7];
  const float* bq      = (const float*)d_in[8];
  const float* Wk      = (const float*)d_in[9];
  const float* bk      = (const float*)d_in[10];
  const float* Wv      = (const float*)d_in[11];
  const float* bv      = (const float*)d_in[12];
  const float* We      = (const float*)d_in[13];
  const float* be      = (const float*)d_in[14];
  const float* snorm_g = (const float*)d_in[15];
  const float* snorm_b = (const float*)d_in[16];
  const float* Wo      = (const float*)d_in[17];
  const float* bo      = (const float*)d_in[18];
  float* out = (float*)d_out;

  // Workspace layout (256B aligned)
  char* ws = (char*)d_ws;
  size_t off = 0;
  auto alloc = [&](size_t bytes) {
    char* p = ws + off;
    off += (bytes + 255) & ~(size_t)255;
    return p;
  };
  const size_t BT = (size_t)B_ * T_;
  const size_t BN = (size_t)B_ * N_;
  half_t* xh  = (half_t*)alloc(BT * D_ * 2);        // LN(x) f16; reused as hsty later
  half_t* xfh = (half_t*)alloc(BN * L_ * 2);        // LN(xf) f16
  half_t* Wqt = (half_t*)alloc((size_t)D_ * D_ * 2);
  half_t* Wkt = (half_t*)alloc((size_t)D_ * L_ * 2);
  half_t* Wvt = (half_t*)alloc((size_t)D_ * L_ * 2);
  half_t* Wot = (half_t*)alloc((size_t)D_ * D_ * 2);
  half_t* qh  = (half_t*)alloc(BT * D_ * 2);
  half_t* khb = (half_t*)alloc(BN * D_ * 2);
  half_t* vhb = (half_t*)alloc(BN * D_ * 2);
  half_t* yb  = (half_t*)alloc(BT * D_ * 2);
  float*  eo  = (float*)alloc((size_t)B_ * 2 * D_ * 4);

  // 1) weights -> f16, transposed (N x K)
  w_to_f16_t<<<4096, 256, 0, stream>>>(Wq, Wqt, D_, D_);
  w_to_f16_t<<<3072, 256, 0, stream>>>(Wk, Wkt, L_, D_);
  w_to_f16_t<<<3072, 256, 0, stream>>>(Wv, Wvt, L_, D_);
  w_to_f16_t<<<4096, 256, 0, stream>>>(Wo, Wot, D_, D_);

  // 2) LayerNorms -> f16
  ln_rows_f16<<<(int)BT, 256, 0, stream>>>(x, norm_g, norm_b, xh, D_);
  ln_rows_f16<<<(int)BN, 256, 0, stream>>>(xf, tnorm_g, tnorm_b, xfh, L_);

  // 3) Projections (Q folds 1/sqrt(dh)=0.125)
  gemm_wmma<false><<<dim3(D_ / 128, (int)(BT / 128)), 256, 0, stream>>>(
      xh, Wqt, bq, nullptr, qh, nullptr, (int)BT, D_, D_, 0.125f);
  gemm_wmma<false><<<dim3(D_ / 128, (int)(BN / 128)), 256, 0, stream>>>(
      xfh, Wkt, bk, nullptr, khb, nullptr, (int)BN, D_, L_, 1.0f);
  gemm_wmma<false><<<dim3(D_ / 128, (int)(BN / 128)), 256, 0, stream>>>(
      xfh, Wvt, bv, nullptr, vhb, nullptr, (int)BN, D_, L_, 1.0f);

  // 4) emb MLP (tiny)
  emb_out_kernel<<<dim3(2 * D_ / 256, B_), 256, 0, stream>>>(emb, We, be, eo);

  // 5) attention: grid (B*H, T/128)
  attn_kernel<<<dim3(B_ * H_, T_ / 128), 256, 0, stream>>>(qh, khb, vhb, yb);

  // 6) stylize (xh buffer reused as f16 h)
  stylize_kernel<<<(int)BT, 256, 0, stream>>>(yb, eo, snorm_g, snorm_b, xh);

  // 7) out = x + h @ Wo + bo
  gemm_wmma<true><<<dim3(D_ / 128, (int)(BT / 128)), 256, 0, stream>>>(
      xh, Wot, bo, x, nullptr, out, (int)BT, D_, D_, 1.0f);
}
